// GATModel_72653666779819
// MI455X (gfx1250) — matrix operand order repaired and checked by reading, not compiled, not run on hardware
//
#include <hip/hip_runtime.h>
#include <hip/hip_bf16.h>

// ---------------------------------------------------------------------------
// Problem constants (from reference)
// ---------------------------------------------------------------------------
#define NNODES   50000
#define NEDGES   1600000
#define ETOT     (NEDGES + NNODES)   // with self loops
#define INCH     128
#define HIDC     64
#define NHEADS   2
#define OUTC2    8
#define NGRAPHS  64
#define NEGSLOPE 0.2f

typedef __attribute__((ext_vector_type(16))) _Float16 v16h;
typedef __attribute__((ext_vector_type(8)))  _Float16 v8h;
typedef __attribute__((ext_vector_type(8)))  float    v8f;

#define KDIM 128          // K is 128 for both GEMMs
#define LDK  136          // padded halves per LDS row (272B: 16B-aligned, bank-staggered)

// ---------------------------------------------------------------------------
// WMMA GEMM:  C[M x Nout] = A[M x 128] (f32) * W[128 x Nout] (f32)
// A converted f32->f16 into LDS; W loaded transposed (Bs[n][k]) zero-padded
// to BN columns.  Each wave computes TW 16x16 tiles with
// v_wmma_f32_16x16x32_f16 over 4 K-steps.
// ---------------------------------------------------------------------------
template<int BM, int BN>
__global__ __launch_bounds__(256)
void gemm_wmma_kernel(const float* __restrict__ A, const float* __restrict__ W,
                      float* __restrict__ C, int M, int NoutActual)
{
    __shared__ _Float16 As[BM * LDK];
    __shared__ _Float16 Bs[BN * LDK];      // transposed: Bs[n*LDK + k]

    const int tid      = threadIdx.x;
    const int lane     = tid & 31;
    const int wave     = tid >> 5;
    const int rowBlock = blockIdx.x * BM;

    // --- stage A tile (BM x 128), f32 -> f16 ---
    for (int idx = tid; idx < BM * KDIM; idx += 256) {
        int m = idx / KDIM, k = idx - m * KDIM;
        int gm = rowBlock + m;
        float v = (gm < M) ? A[(size_t)gm * KDIM + k] : 0.0f;
        As[m * LDK + k] = (_Float16)v;
    }
    // --- stage B transposed (BN x 128), zero pad cols >= NoutActual ---
    for (int idx = tid; idx < BN * KDIM; idx += 256) {
        int n = idx / KDIM, k = idx - n * KDIM;
        float v = (n < NoutActual) ? W[(size_t)k * NoutActual + n] : 0.0f;
        Bs[n * LDK + k] = (_Float16)v;
    }
    __syncthreads();

    constexpr int CT    = BN / 16;
    constexpr int TOTAL = (BM / 16) * CT;
    constexpr int TW    = TOTAL / 8;            // tiles per wave
    static_assert(TOTAL % 8 == 0 && TW >= 1, "tile mapping");
    const int tileBase = wave * TW;
    const int rt       = tileBase / CT;         // all TW tiles share one row tile

    v8f acc[TW] = {};

    const int half16 = lane >> 4;               // 0 / 1
    const int l15    = lane & 15;

    for (int k0 = 0; k0 < KDIM; k0 += 32) {
        // A fragment (16x32 f16): lane<16 -> K k0+[0..7] & k0+[16..23];
        //                         lane>=16 -> K k0+[8..15] & k0+[24..31]
        const _Float16* pa = &As[(rt * 16 + l15) * LDK + k0 + half16 * 8];
        v8h a_lo = *reinterpret_cast<const v8h*>(pa);
        v8h a_hi = *reinterpret_cast<const v8h*>(pa + 16);
        v16h afrag;
        #pragma unroll
        for (int i = 0; i < 8; ++i) { afrag[i] = a_lo[i]; afrag[8 + i] = a_hi[i]; }

        #pragma unroll
        for (int t = 0; t < TW; ++t) {
            const int ct = (tileBase + t) % CT;
            // B fragment (32x16 f16): lane<16 -> col l15, K k0+[0..7] & k0+[16..23];
            //                         lane>=16 -> col l15, K k0+[8..15] & k0+[24..31]
            const _Float16* pb = &Bs[(ct * 16 + l15) * LDK + k0 + half16 * 8];
            v8h b_lo = *reinterpret_cast<const v8h*>(pb);
            v8h b_hi = *reinterpret_cast<const v8h*>(pb + 16);
            v16h bfrag;
            #pragma unroll
            for (int i = 0; i < 8; ++i) { bfrag[i] = b_lo[i]; bfrag[8 + i] = b_hi[i]; }

            acc[t] = __builtin_amdgcn_wmma_f32_16x16x32_f16(
                         false, afrag, false, bfrag, (short)0, acc[t], false, false);
        }
    }

    // --- store C: acc[t][r] -> row rt*16 + r + half16*8, col ct*16 + l15 ---
    #pragma unroll
    for (int t = 0; t < TW; ++t) {
        const int ct = (tileBase + t) % CT;
        #pragma unroll
        for (int r = 0; r < 8; ++r) {
            int gm = rowBlock + rt * 16 + r + half16 * 8;
            int gn = ct * 16 + l15;
            if (gm < M && gn < NoutActual)
                C[(size_t)gm * NoutActual + gn] = acc[t][r];
        }
    }
}

// ---------------------------------------------------------------------------
// Elementwise / edge kernels
// ---------------------------------------------------------------------------
__global__ __launch_bounds__(256)
void fill_kernel(float* __restrict__ p, int n, float v)
{
    int i = blockIdx.x * blockDim.x + threadIdx.x;
    if (i < n) p[i] = v;
}

// per (node, head): a_src/a_dst = <h[node,head,:], att[head,:]>
__global__ __launch_bounds__(256)
void attn_coef_kernel(const float* __restrict__ h,
                      const float* __restrict__ att_src,
                      const float* __restrict__ att_dst,
                      float* __restrict__ asrc, float* __restrict__ adst,
                      int n, int H, int C)
{
    int i = blockIdx.x * blockDim.x + threadIdx.x;
    if (i >= n * H) return;
    int node = i / H, hd = i - node * H;
    const float* hp = h + (size_t)node * H * C + hd * C;
    float s = 0.0f, d = 0.0f;
    for (int c = 0; c < C; ++c) {
        float v = hp[c];
        s += v * att_src[hd * C + c];
        d += v * att_dst[hd * C + c];
    }
    asrc[i] = s;
    adst[i] = d;
}

__device__ inline void atomicMaxF(float* addr, float val)
{
    int* ia  = (int*)addr;
    int  old = *ia;
    while (__int_as_float(old) < val) {
        int assumed = old;
        old = atomicCAS(ia, assumed, __float_as_int(val));
        if (old == assumed) break;
    }
}

__device__ inline void edge_src_dst(const int* __restrict__ ei, int e, int nE,
                                    int& s, int& d)
{
    if (e < nE) { s = ei[e]; d = ei[nE + e]; }
    else        { s = d = e - nE; }            // self loops appended
}

// pass 1: logits + segment max over dst
__global__ __launch_bounds__(256)
void edge_logits_max_kernel(const int* __restrict__ ei, int nE, int H,
                            const float* __restrict__ asrc,
                            const float* __restrict__ adst,
                            float* __restrict__ elog, float* __restrict__ mmax,
                            int total)
{
    int i = blockIdx.x * blockDim.x + threadIdx.x;
    if (i >= total) return;
    int e = i / H, hd = i - e * H;
    int s, d;
    edge_src_dst(ei, e, nE, s, d);
    float l = asrc[s * H + hd] + adst[d * H + hd];
    l = (l >= 0.0f) ? l : NEGSLOPE * l;        // leaky relu
    elog[i] = l;
    atomicMaxF(&mmax[d * H + hd], l);
}

// pass 2: e = exp(l - m[dst]); segment sum of e
__global__ __launch_bounds__(256)
void edge_exp_kernel(const int* __restrict__ ei, int nE, int H,
                     float* __restrict__ elog,
                     const float* __restrict__ mmax, float* __restrict__ den,
                     int total)
{
    int i = blockIdx.x * blockDim.x + threadIdx.x;
    if (i >= total) return;
    int e = i / H, hd = i - e * H;
    int s, d;
    edge_src_dst(ei, e, nE, s, d);
    float ex = __expf(elog[i] - mmax[d * H + hd]);
    elog[i] = ex;
    atomicAdd(&den[d * H + hd], ex);
}

// pass 3: out[dst] += h[src] * alpha   (vectorized float4 gathers)
__global__ __launch_bounds__(256)
void edge_agg_kernel(const int* __restrict__ ei, int nE, int H, int C,
                     const float* __restrict__ h,
                     const float* __restrict__ elog,
                     const float* __restrict__ den,
                     float* __restrict__ out, int total)
{
    int i = blockIdx.x * blockDim.x + threadIdx.x;
    if (i >= total) return;
    int e = i / H, hd = i - e * H;
    int s, d;
    edge_src_dst(ei, e, nE, s, d);
    float alpha = elog[i] / den[d * H + hd];
    const float4* hs = reinterpret_cast<const float4*>(h + (size_t)s * H * C + hd * C);
    float* op = out + (size_t)d * H * C + hd * C;
    for (int c4 = 0; c4 < C / 4; ++c4) {
        float4 v = hs[c4];
        atomicAdd(&op[c4 * 4 + 0], v.x * alpha);
        atomicAdd(&op[c4 * 4 + 1], v.y * alpha);
        atomicAdd(&op[c4 * 4 + 2], v.z * alpha);
        atomicAdd(&op[c4 * 4 + 3], v.w * alpha);
    }
}

// x = relu(x + bias[col])   (in-place, ld = ncols)
__global__ __launch_bounds__(256)
void bias_relu_kernel(float* __restrict__ x, const float* __restrict__ bias,
                      int total, int ncols)
{
    int i = blockIdx.x * blockDim.x + threadIdx.x;
    if (i >= total) return;
    float v = x[i] + bias[i % ncols];
    x[i] = (v > 0.0f) ? v : 0.0f;
}

// mean-pool accumulation: pooled[g] += relu(out2[n]+b2), cnt[g] += 1
__global__ __launch_bounds__(256)
void pool_kernel(const float* __restrict__ out2, const float* __restrict__ b2,
                 const int* __restrict__ batch,
                 float* __restrict__ pooled, float* __restrict__ cnt, int n)
{
    int i = blockIdx.x * blockDim.x + threadIdx.x;
    if (i >= n) return;
    int g = batch[i];
    #pragma unroll
    for (int c = 0; c < OUTC2; ++c) {
        float v = out2[(size_t)i * OUTC2 + c] + b2[c];
        v = (v > 0.0f) ? v : 0.0f;
        atomicAdd(&pooled[g * OUTC2 + c], v);
    }
    atomicAdd(&cnt[g], 1.0f);
}

__global__ void final_kernel(const float* __restrict__ pooled,
                             const float* __restrict__ cnt,
                             const float* __restrict__ fcW,
                             const float* __restrict__ fcb,
                             float* __restrict__ y)
{
    int g = threadIdx.x;
    if (g >= NGRAPHS) return;
    float c = cnt[g];
    c = (c > 1.0f) ? c : 1.0f;
    float s = fcb[0];
    #pragma unroll
    for (int k = 0; k < OUTC2; ++k)
        s += (pooled[g * OUTC2 + k] / c) * fcW[k];
    y[g] = 2.0f / (1.0f + __expf(-s)) - 1.0f;
}

// ---------------------------------------------------------------------------
// Launch
// ---------------------------------------------------------------------------
static inline int cdiv(long long a, int b) { return (int)((a + b - 1) / b); }

extern "C" void kernel_launch(void* const* d_in, const int* in_sizes, int n_in,
                              void* d_out, int out_size, void* d_ws, size_t ws_size,
                              hipStream_t stream)
{
    const float* x     = (const float*)d_in[0];
    const int*   ei    = (const int*)  d_in[1];
    const int*   batch = (const int*)  d_in[2];
    const float* W1    = (const float*)d_in[3];
    const float* as1   = (const float*)d_in[4];
    const float* ad1   = (const float*)d_in[5];
    const float* b1    = (const float*)d_in[6];
    const float* W2    = (const float*)d_in[7];
    const float* as2   = (const float*)d_in[8];
    const float* ad2   = (const float*)d_in[9];
    const float* b2    = (const float*)d_in[10];
    const float* fcW   = (const float*)d_in[11];
    const float* fcb   = (const float*)d_in[12];
    float* y = (float*)d_out;

    // workspace carve-up (256B aligned)
    char* wp = (char*)d_ws;
    auto carve = [&](size_t bytes) -> float* {
        float* p = (float*)wp;
        wp += (bytes + 255) & ~(size_t)255;
        return p;
    };
    float* h1    = carve((size_t)NNODES * 128 * 4);   // layer1 pre-agg features
    float* out1  = carve((size_t)NNODES * 128 * 4);   // layer1 agg (then x2 in-place)
    float* asrc1 = carve((size_t)NNODES * 2 * 4);
    float* adst1 = carve((size_t)NNODES * 2 * 4);
    float* m1    = carve((size_t)NNODES * 2 * 4);
    float* den1  = carve((size_t)NNODES * 2 * 4);
    float* elog1 = carve((size_t)ETOT * 2 * 4);
    float* h2    = carve((size_t)NNODES * OUTC2 * 4);
    float* out2  = carve((size_t)NNODES * OUTC2 * 4);
    float* asrc2 = carve((size_t)NNODES * 4);
    float* adst2 = carve((size_t)NNODES * 4);
    float* m2    = carve((size_t)NNODES * 4);
    float* den2  = carve((size_t)NNODES * 4);
    float* elog2 = carve((size_t)ETOT * 4);
    float* pooled= carve((size_t)NGRAPHS * OUTC2 * 4);
    float* cnt   = carve((size_t)NGRAPHS * 4);

    const int B = 256;

    // init accumulators (must happen every call: graph replay, ws not re-poisoned)
    fill_kernel<<<cdiv(NNODES * 2, B), B, 0, stream>>>(m1,   NNODES * 2, -3.0e38f);
    fill_kernel<<<cdiv(NNODES * 2, B), B, 0, stream>>>(den1, NNODES * 2, 0.0f);
    fill_kernel<<<cdiv((long long)NNODES * 128, B), B, 0, stream>>>(out1, NNODES * 128, 0.0f);
    fill_kernel<<<cdiv(NNODES, B), B, 0, stream>>>(m2,   NNODES, -3.0e38f);
    fill_kernel<<<cdiv(NNODES, B), B, 0, stream>>>(den2, NNODES, 0.0f);
    fill_kernel<<<cdiv(NNODES * OUTC2, B), B, 0, stream>>>(out2, NNODES * OUTC2, 0.0f);
    fill_kernel<<<1, B, 0, stream>>>(pooled, NGRAPHS * OUTC2, 0.0f);
    fill_kernel<<<1, B, 0, stream>>>(cnt, NGRAPHS, 0.0f);

    // ---- layer 1 ----
    gemm_wmma_kernel<64, 128><<<cdiv(NNODES, 64), B, 0, stream>>>(x, W1, h1, NNODES, 128);
    attn_coef_kernel<<<cdiv(NNODES * 2, B), B, 0, stream>>>(h1, as1, ad1, asrc1, adst1,
                                                            NNODES, NHEADS, HIDC);
    {
        int total = ETOT * NHEADS;
        edge_logits_max_kernel<<<cdiv(total, B), B, 0, stream>>>(ei, NEDGES, NHEADS,
                                                                 asrc1, adst1, elog1, m1, total);
        edge_exp_kernel<<<cdiv(total, B), B, 0, stream>>>(ei, NEDGES, NHEADS,
                                                          elog1, m1, den1, total);
        edge_agg_kernel<<<cdiv(total, B), B, 0, stream>>>(ei, NEDGES, NHEADS, HIDC,
                                                          h1, elog1, den1, out1, total);
    }
    bias_relu_kernel<<<cdiv((long long)NNODES * 128, B), B, 0, stream>>>(out1, b1,
                                                                         NNODES * 128, 128);

    // ---- layer 2 ----
    gemm_wmma_kernel<128, 16><<<cdiv(NNODES, 128), B, 0, stream>>>(out1, W2, h2, NNODES, OUTC2);
    attn_coef_kernel<<<cdiv(NNODES, B), B, 0, stream>>>(h2, as2, ad2, asrc2, adst2,
                                                        NNODES, 1, OUTC2);
    {
        int total = ETOT;
        edge_logits_max_kernel<<<cdiv(total, B), B, 0, stream>>>(ei, NEDGES, 1,
                                                                 asrc2, adst2, elog2, m2, total);
        edge_exp_kernel<<<cdiv(total, B), B, 0, stream>>>(ei, NEDGES, 1,
                                                          elog2, m2, den2, total);
        edge_agg_kernel<<<cdiv(total, B), B, 0, stream>>>(ei, NEDGES, 1, OUTC2,
                                                          h2, elog2, den2, out2, total);
    }

    // ---- pool + fc + sigmoid ----
    pool_kernel<<<cdiv(NNODES, B), B, 0, stream>>>(out2, b2, batch, pooled, cnt, NNODES);
    final_kernel<<<1, 64, 0, stream>>>(pooled, cnt, fcW, fcb, y);
}
